// Intersection_76295799046217
// MI455X (gfx1250) — matrix operand order repaired and checked by reading, not compiled, not run on hardware
//
#include <hip/hip_runtime.h>
#include <hip/hip_bf16.h>
#include <math.h>

// Pipeline (fp32 in/out, bf16 WMMA compute, fp32 accumulate):
//  1) w    = ew @ Ww0^T + bw0            [E,512]   (WMMA GEMM, async-LDS A stage)
//  2) out  = x  @ W0^T  + b0             [N,512]   (WMMA GEMM)
//  3) out += DAMPING * segment_sum(w,src)          (atomic scatter)
//  4) node_scalar = rowsum(out); rn = 1/||out||    (wave reductions)
//  5) w += DAMPING*node_scalar[src]; fused BN stats
//  6) BN scale/shift from stats
//  7) attention pass1: streaming row max/sum-exp; Q frags resident in VGPRs,
//     K tile converted once to bf16 in LDS and shared by all 8 waves
//  8) attention pass2: out2 = softmax(sim) @ out; V tile staged in LDS in both
//     row-major (for S) and transposed (for P@V vector loads) bf16 layouts
//  9) d_out[0:N*VD]  = out2 @ W1^T + b1            (WMMA GEMM)
// 10) d_out[N*VD: ]  = BN(w) @ Ww1^T + bw1         (WMMA GEMM, BN fused in A-load)
//
// Workspace: ~545 MB (w 512MB + out/out2 16MB each + small vectors).

#define N_NODES 8192
#define N_EDGES 262144
#define VDIM    256
#define EDIM    256
#define HDIM    512
#define DAMP    0.45f
#define BN_EPS  1e-5f
#define EB      256   // edges per block in w_update_stats

typedef __attribute__((ext_vector_type(16))) __bf16 v16bf;
typedef __attribute__((ext_vector_type(8)))  __bf16 v8bf;
typedef __attribute__((ext_vector_type(8)))  float  v8f;
typedef int v4i_t __attribute__((vector_size(16)));   // matches builtin param type

#if defined(__has_builtin)
#if __has_builtin(__builtin_amdgcn_global_load_async_to_lds_b128)
#define USE_ASYNC_LDS 1
#endif
#endif

// 16-byte global -> LDS copy; async (ASYNCcnt-tracked) when available.
__device__ __forceinline__ void copy16_g2l(const float* gp, float* lp) {
#ifdef USE_ASYNC_LDS
  __builtin_amdgcn_global_load_async_to_lds_b128(
      (__attribute__((address_space(1))) v4i_t*)
          (const __attribute__((address_space(1))) void*)gp,
      (__attribute__((address_space(3))) v4i_t*)
          (__attribute__((address_space(3))) void*)lp,
      0, 0);
#else
  *(float4*)lp = *(const float4*)gp;
#endif
}

__device__ __forceinline__ void wait_async_lds() {
#ifdef USE_ASYNC_LDS
  asm volatile("s_wait_asynccnt 0x0" ::: "memory");
#endif
}

// ---------------- WMMA fragment loaders (wave32, ISA layouts) ----------------

// A (16x32 bf16, MxK) from global fp32: lane<16 holds row M=lane,
// K = k0..k0+7 / k0+16..k0+23; lane>=16 holds K = k0+8..15 / k0+24..31.
__device__ __forceinline__ v16bf load_a_frag(const float* __restrict__ A, int ld,
                                             int row0, int k0, int lane) {
  int r  = row0 + (lane & 15);
  int kb = k0 + ((lane >> 4) << 3);
  const float* p = A + (size_t)r * ld + kb;
  v16bf a;
#pragma unroll
  for (int i = 0; i < 8; ++i) a[i] = (__bf16)p[i];
#pragma unroll
  for (int i = 0; i < 8; ++i) a[8 + i] = (__bf16)p[16 + i];
  return a;
}

// A fragment from a 16x32 fp32 tile staged in LDS, with optional per-K affine
// (BatchNorm fused). k0 is the global K offset for the affine vectors.
__device__ __forceinline__ v16bf load_a_frag_lds(const float* Ash, int k0, int lane,
                                                 const float* __restrict__ ascale,
                                                 const float* __restrict__ ashift) {
  int r  = lane & 15;
  int kb = (lane >> 4) << 3;
  const float* p = Ash + r * 32 + kb;
  v16bf a;
  if (ascale) {
    int kg = k0 + kb;
#pragma unroll
    for (int i = 0; i < 8; ++i)
      a[i] = (__bf16)(p[i] * ascale[kg + i] + ashift[kg + i]);
#pragma unroll
    for (int i = 0; i < 8; ++i)
      a[8 + i] = (__bf16)(p[16 + i] * ascale[kg + 16 + i] + ashift[kg + 16 + i]);
  } else {
#pragma unroll
    for (int i = 0; i < 8; ++i) a[i] = (__bf16)p[i];
#pragma unroll
    for (int i = 0; i < 8; ++i) a[8 + i] = (__bf16)p[16 + i];
  }
  return a;
}

// B (32x16 bf16, KxN) for NT GEMM from global fp32 W row-major [Nc,K]:
// lane<16: N=lane, K=k0..k0+7 / k0+16..23; lane>=16: N=lane-16, K=k0+8..15 / k0+24..31.
__device__ __forceinline__ v16bf load_b_frag_nt(const float* __restrict__ W, int ld,
                                                int n0, int k0, int lane) {
  int n  = n0 + (lane & 15);
  int kb = k0 + ((lane >> 4) << 3);
  const float* p = W + (size_t)n * ld + kb;
  v16bf b;
#pragma unroll
  for (int i = 0; i < 8; ++i) b[i] = (__bf16)p[i];
#pragma unroll
  for (int i = 0; i < 8; ++i) b[8 + i] = (__bf16)p[16 + i];
  return b;
}

// B (32x16) for NT GEMM from a bf16 row-major tile in LDS (ld elements/row).
__device__ __forceinline__ v16bf load_b_frag_lds_nt(const __bf16* T, int ld,
                                                    int nbase, int k0, int lane) {
  int n  = nbase + (lane & 15);
  int kb = k0 + ((lane >> 4) << 3);
  const __bf16* p = T + (size_t)n * ld + kb;
  v8bf lo = *(const v8bf*)p;          // K = kb..kb+7
  v8bf hv = *(const v8bf*)(p + 16);   // K = kb+16..kb+23
  return __builtin_shufflevector(lo, hv, 0, 1, 2, 3, 4, 5, 6, 7, 8, 9, 10, 11, 12, 13, 14, 15);
}

// ---------------- Generic NT GEMM ---------------------------------------------
// C[M,Nc] = (affine? A) [M,K] @ B[Nc,K]^T + bias.  Block owns one 16-row A
// strip (staged per k-step into LDS with async load-to-LDS); each wave owns 64
// output columns (4 accumulators, A fragment reused 4x). blockDim.x == Nc/2.

__global__ void gemm_nt_wmma(const float* __restrict__ A, const float* __restrict__ B,
                             const float* __restrict__ bias,
                             const float* __restrict__ ascale,
                             const float* __restrict__ ashift,
                             float* __restrict__ C, int M, int Nc, int K) {
  __shared__ __attribute__((aligned(16))) float Ash[16 * 32];
  int tid  = threadIdx.x;
  int lane = tid & 31;
  int wave = tid >> 5;
  int row0 = blockIdx.x << 4;
  int n0   = wave << 6;                  // 64 columns per wave
  v8f acc[4] = {};
  for (int k0 = 0; k0 < K; k0 += 32) {
    if (tid < 128) {                     // stage 16x32 fp32 A tile (2KB)
      int rr  = tid >> 3;
      int seg = (tid & 7) << 2;
      copy16_g2l(A + (size_t)(row0 + rr) * K + k0 + seg, &Ash[rr * 32 + seg]);
    }
    wait_async_lds();
    __syncthreads();
    v16bf a = load_a_frag_lds(Ash, k0, lane, ascale, ashift);
#pragma unroll
    for (int t = 0; t < 4; ++t) {
      v16bf b = load_b_frag_nt(B, K, n0 + (t << 4), k0, lane);
      acc[t] = __builtin_amdgcn_wmma_f32_16x16x32_bf16(false, a, false, b,
                                                       (short)0, acc[t], false, false);
    }
    __syncthreads();
  }
  int r = row0 + ((lane >> 4) << 3);
#pragma unroll
  for (int t = 0; t < 4; ++t) {
    int c = n0 + (t << 4) + (lane & 15);
    float bi = bias ? bias[c] : 0.0f;
#pragma unroll
    for (int v = 0; v < 8; ++v)
      C[(size_t)(r + v) * Nc + c] = acc[t][v] + bi;
  }
}

// ---------------- out += DAMPING * segment_sum(w, src) -------------------------

__global__ void scatter_add_kernel(const float* __restrict__ w,
                                   const int* __restrict__ src,
                                   float* __restrict__ out) {
  size_t idx = (size_t)blockIdx.x * blockDim.x + threadIdx.x;
  size_t total = (size_t)N_EDGES * (HDIM / 4);
  if (idx >= total) return;
  size_t e = idx / (HDIM / 4);
  int h = (int)(idx % (HDIM / 4)) * 4;
  const float4 wv = *(const float4*)(w + e * HDIM + h);
  int s = src[e];
  float* o = out + (size_t)s * HDIM + h;
  atomicAdd(o + 0, DAMP * wv.x);
  atomicAdd(o + 1, DAMP * wv.y);
  atomicAdd(o + 2, DAMP * wv.z);
  atomicAdd(o + 3, DAMP * wv.w);
}

// ---------------- per-row sum and 1/||row|| (one wave32 per row) ---------------

__global__ void row_stats_kernel(const float* __restrict__ out,
                                 float* __restrict__ nscal,
                                 float* __restrict__ rn) {
  int row  = blockIdx.x * (blockDim.x >> 5) + (threadIdx.x >> 5);
  int lane = threadIdx.x & 31;
  if (row >= N_NODES) return;
  const float* p = out + (size_t)row * HDIM;
  float s = 0.f, q = 0.f;
  for (int h = lane; h < HDIM; h += 32) { float v = p[h]; s += v; q += v * v; }
#pragma unroll
  for (int off = 16; off > 0; off >>= 1) {
    s += __shfl_xor(s, off);
    q += __shfl_xor(q, off);
  }
  if (lane == 0) { nscal[row] = s; rn[row] = rsqrtf(q); }
}

// ---------------- w += DAMPING*nscal[src], fused BN-stat accumulation ----------

__global__ void w_update_stats_kernel(float* __restrict__ w,
                                      const int* __restrict__ src,
                                      const float* __restrict__ nscal,
                                      float* __restrict__ s1,
                                      float* __restrict__ s2) {
  int c = threadIdx.x;                 // blockDim.x == HDIM, coalesced channel
  size_t e0 = (size_t)blockIdx.x * EB;
  float a1 = 0.f, a2 = 0.f;
  for (int i = 0; i < EB; ++i) {
    size_t e = e0 + i;
    float add = DAMP * nscal[src[e]];
    float v = w[e * HDIM + c] + add;
    w[e * HDIM + c] = v;
    a1 += v; a2 += v * v;
  }
  atomicAdd(&s1[c], a1);
  atomicAdd(&s2[c], a2);
}

__global__ void bn_finalize_kernel(const float* __restrict__ s1,
                                   const float* __restrict__ s2,
                                   const float* __restrict__ gamma,
                                   const float* __restrict__ beta,
                                   float* __restrict__ scale,
                                   float* __restrict__ shift) {
  int c = threadIdx.x;
  float mean = s1[c] * (1.0f / N_EDGES);
  float var  = s2[c] * (1.0f / N_EDGES) - mean * mean;
  float r = rsqrtf(var + BN_EPS);
  float g = gamma[c] * r;
  scale[c] = g;
  shift[c] = beta[c] - mean * g;
}

__global__ void zero_kernel(float* __restrict__ p, int n) {
  int i = blockIdx.x * blockDim.x + threadIdx.x;
  if (i < n) p[i] = 0.f;
}

// ---------------- attention pass 1: streaming row max / sum-exp ----------------
// sim = (out @ out^T) * rn_i * rn_j - I.  8 waves x one 16-query tile each.
// Q fragments live in VGPRs (converted once); the 16-key tile is converted to
// bf16 in LDS once per block and shared by all 8 waves.

__global__ void attn_pass1_kernel(const float* __restrict__ out,
                                  const float* __restrict__ rn,
                                  float* __restrict__ rowm,
                                  float* __restrict__ rowl) {
  __shared__ __attribute__((aligned(32))) __bf16 Ksh[16 * HDIM];   // 16KB
  int tid  = threadIdx.x;
  int lane = tid & 31;
  int wave = tid >> 5;
  int q0 = (blockIdx.x * 8 + wave) << 4;
  int hi = lane >> 4;
  v16bf qa[16];
#pragma unroll
  for (int kt = 0; kt < 16; ++kt)
    qa[kt] = load_a_frag(out, HDIM, q0, kt << 5, lane);
  float rnq[8], rm[8], rl[8];
#pragma unroll
  for (int v = 0; v < 8; ++v) {
    rnq[v] = rn[q0 + v + hi * 8];
    rm[v] = -__builtin_inff();
    rl[v] = 0.f;
  }
  for (int j0 = 0; j0 < N_NODES; j0 += 16) {
    __syncthreads();
    for (int idx = tid; idx < 16 * HDIM; idx += 256)
      Ksh[idx] = (__bf16)out[(size_t)(j0 + (idx >> 9)) * HDIM + (idx & (HDIM - 1))];
    __syncthreads();
    v8f s = {};
#pragma unroll
    for (int kt = 0; kt < 16; ++kt) {
      v16bf b = load_b_frag_lds_nt(Ksh, HDIM, 0, kt << 5, lane);
      s = __builtin_amdgcn_wmma_f32_16x16x32_bf16(false, qa[kt], false, b,
                                                  (short)0, s, false, false);
    }
    int col = j0 + (lane & 15);
    float rnc = rn[col];
#pragma unroll
    for (int v = 0; v < 8; ++v) {
      int rowi = q0 + v + hi * 8;
      float sv = s[v] * rnq[v] * rnc;
      if (rowi == col) sv -= 1.0f;        // minus identity
      float mx = sv;
#pragma unroll
      for (int off = 8; off > 0; off >>= 1) mx = fmaxf(mx, __shfl_xor(mx, off));
      float mnew = fmaxf(rm[v], mx);
      float p = __expf(sv - mnew);
      float ps = p;
#pragma unroll
      for (int off = 8; off > 0; off >>= 1) ps += __shfl_xor(ps, off);
      rl[v] = rl[v] * __expf(rm[v] - mnew) + ps;
      rm[v] = mnew;
    }
  }
  if ((lane & 15) == 0) {
#pragma unroll
    for (int v = 0; v < 8; ++v) {
      int rowi = q0 + v + hi * 8;
      rowm[rowi] = rm[v];
      rowl[rowi] = rl[v];
    }
  }
}

// ---------------- attention pass 2: out2 = softmax(sim) @ out ------------------
// Block = 8 waves, one 16-query tile.  V tile (32 keys x 512) is converted to
// bf16 once per step into BOTH a row-major copy Vsh (K-contiguous, feeds the S
// GEMM of waves 0,1) and a transposed copy VshT[col][key] (key-contiguous, so
// the P@V B-fragment is two aligned 16B vector LDS loads per lane).

__global__ void attn_pass2_kernel(const float* __restrict__ out,
                                  const float* __restrict__ rn,
                                  const float* __restrict__ rowm,
                                  const float* __restrict__ rowl,
                                  float* __restrict__ out2) {
  __shared__ __attribute__((aligned(32))) __bf16 Vsh [32 * HDIM];  // 32KB row-major
  __shared__ __attribute__((aligned(32))) __bf16 VshT[HDIM * 32];  // 32KB transposed
  __shared__ __attribute__((aligned(32))) __bf16 Psh[16 * 32];     // 1KB
  int tid  = threadIdx.x;
  int lane = tid & 31;
  int wave = tid >> 5;
  int q0 = blockIdx.x << 4;
  int hc = wave << 6;                   // 8 waves x 64 cols = 512
  int hi = lane >> 4;
  v16bf qa[16];
  float rnq[8], rm[8], rli[8];
  if (wave < 2) {                       // wave-uniform
#pragma unroll
    for (int kt = 0; kt < 16; ++kt)
      qa[kt] = load_a_frag(out, HDIM, q0, kt << 5, lane);
#pragma unroll
    for (int v = 0; v < 8; ++v) {
      int rowi = q0 + v + hi * 8;
      rnq[v] = rn[rowi];
      rm[v]  = rowm[rowi];
      rli[v] = 1.0f / rowl[rowi];
    }
  }
  v8f acc[4] = {};
  for (int j0 = 0; j0 < N_NODES; j0 += 32) {
    __syncthreads();
    for (int idx = tid; idx < 32 * HDIM; idx += 256) {
      int key = idx >> 9;
      int col = idx & (HDIM - 1);
      __bf16 v = (__bf16)out[(size_t)(j0 + key) * HDIM + col];
      Vsh[idx] = v;
      VshT[col * 32 + key] = v;
    }
    __syncthreads();
    if (wave < 2) {                     // build one 16x16 P tile per wave
      int jb = j0 + (wave << 4);
      v8f s = {};
#pragma unroll
      for (int kt = 0; kt < 16; ++kt) {
        v16bf b = load_b_frag_lds_nt(Vsh, HDIM, wave << 4, kt << 5, lane);
        s = __builtin_amdgcn_wmma_f32_16x16x32_bf16(false, qa[kt], false, b,
                                                    (short)0, s, false, false);
      }
      int col = jb + (lane & 15);
      float rnc = rn[col];
      int kloc = (wave << 4) + (lane & 15);
#pragma unroll
      for (int v = 0; v < 8; ++v) {
        int rowi = q0 + v + hi * 8;
        float sv = s[v] * rnq[v] * rnc;
        if (rowi == col) sv -= 1.0f;
        float p = __expf(sv - rm[v]) * rli[v];
        Psh[(v + hi * 8) * 32 + kloc] = (__bf16)p;
      }
    }
    __syncthreads();
    {
      // P (16x32 bf16 row-major in LDS) as A fragment.
      int r  = lane & 15;
      int kb = hi << 3;
      const __bf16* pp = &Psh[r * 32 + kb];
      v16bf a;
#pragma unroll
      for (int i = 0; i < 8; ++i) a[i] = pp[i];
#pragma unroll
      for (int i = 0; i < 8; ++i) a[8 + i] = pp[16 + i];
#pragma unroll
      for (int t = 0; t < 4; ++t) {
        // B[k][n] = VshT[hc + t*16 + n][k]; K-contiguous -> two 16B loads.
        v16bf b = load_b_frag_lds_nt(VshT, 32, hc + (t << 4), 0, lane);
        acc[t] = __builtin_amdgcn_wmma_f32_16x16x32_bf16(false, a, false, b,
                                                         (short)0, acc[t], false, false);
      }
    }
  }
  int r = q0 + ((lane >> 4) << 3);
#pragma unroll
  for (int t = 0; t < 4; ++t) {
    int c = hc + (t << 4) + (lane & 15);
#pragma unroll
    for (int v = 0; v < 8; ++v)
      out2[(size_t)(r + v) * HDIM + c] = acc[t][v];
  }
}

// ---------------- launch ------------------------------------------------------

extern "C" void kernel_launch(void* const* d_in, const int* in_sizes, int n_in,
                              void* d_out, int out_size, void* d_ws, size_t ws_size,
                              hipStream_t stream) {
  (void)in_sizes; (void)n_in; (void)out_size; (void)ws_size;
  const float* x    = (const float*)d_in[0];
  const float* ew   = (const float*)d_in[1];
  const int*   eidx = (const int*)  d_in[2];
  const float* W0   = (const float*)d_in[3];
  const float* b0   = (const float*)d_in[4];
  const float* Ww0  = (const float*)d_in[5];
  const float* bw0  = (const float*)d_in[6];
  const float* W1   = (const float*)d_in[7];
  const float* b1   = (const float*)d_in[8];
  const float* Ww1  = (const float*)d_in[9];
  const float* bw1  = (const float*)d_in[10];
  const float* gamma = (const float*)d_in[11];
  const float* beta  = (const float*)d_in[12];
  const int* src = eidx;               // edge_index[0] = first E ints

  float* base  = (float*)d_ws;
  float* w     = base;                               // E*HD
  float* outb  = w    + (size_t)N_EDGES * HDIM;      // N*HD
  float* out2  = outb + (size_t)N_NODES * HDIM;      // N*HD
  float* nscal = out2 + (size_t)N_NODES * HDIM;      // N
  float* rn    = nscal + N_NODES;                    // N
  float* rowm  = rn    + N_NODES;                    // N
  float* rowl  = rowm  + N_NODES;                    // N
  float* s1    = rowl  + N_NODES;                    // HD
  float* s2    = s1 + HDIM;                          // HD (contiguous with s1)
  float* scale = s2 + HDIM;                          // HD
  float* shift = scale + HDIM;                       // HD

  float* dout1 = (float*)d_out;                      // [N, VD]
  float* dout2 = dout1 + (size_t)N_NODES * VDIM;     // [E, ED]

  // 1) w = ew @ Ww0^T + bw0            (blockDim = Nc/2)
  gemm_nt_wmma<<<dim3(N_EDGES / 16), dim3(HDIM / 2), 0, stream>>>(
      ew, Ww0, bw0, nullptr, nullptr, w, N_EDGES, HDIM, EDIM);
  // 2) out = x @ W0^T + b0
  gemm_nt_wmma<<<dim3(N_NODES / 16), dim3(HDIM / 2), 0, stream>>>(
      x, W0, b0, nullptr, nullptr, outb, N_NODES, HDIM, VDIM);
  // 3) out += DAMPING * segment_sum(w, src)
  {
    size_t total = (size_t)N_EDGES * (HDIM / 4);
    scatter_add_kernel<<<dim3((unsigned)((total + 255) / 256)), 256, 0, stream>>>(
        w, src, outb);
  }
  // 4) node_scalar + reciprocal norms
  row_stats_kernel<<<dim3(N_NODES / 8), 256, 0, stream>>>(outb, nscal, rn);
  // 5) zero BN accumulators (s1,s2 contiguous)
  zero_kernel<<<dim3(4), 256, 0, stream>>>(s1, 2 * HDIM);
  // 6) w += DAMPING*nscal[src]  (fused BN stats)
  w_update_stats_kernel<<<dim3(N_EDGES / EB), dim3(HDIM), 0, stream>>>(
      w, src, nscal, s1, s2);
  // 7) BN scale/shift
  bn_finalize_kernel<<<dim3(1), dim3(HDIM), 0, stream>>>(s1, s2, gamma, beta,
                                                         scale, shift);
  // 8) attention pass 1 (row max / sum-exp)
  attn_pass1_kernel<<<dim3(N_NODES / 128), 256, 0, stream>>>(outb, rn, rowm, rowl);
  // 9) attention pass 2 (weighted accumulate)
  attn_pass2_kernel<<<dim3(N_NODES / 16), 256, 0, stream>>>(outb, rn, rowm, rowl,
                                                            out2);
  // 10) d_out part 1: out2 @ W1^T + b1
  gemm_nt_wmma<<<dim3(N_NODES / 16), dim3(VDIM / 2), 0, stream>>>(
      out2, W1, b1, nullptr, nullptr, dout1, N_NODES, VDIM, HDIM);
  // 11) d_out part 2: BN(w) @ Ww1^T + bw1  (BN fused into A-operand load)
  gemm_nt_wmma<<<dim3(N_EDGES / 16), dim3(EDIM / 2), 0, stream>>>(
      w, Ww1, bw1, scale, shift, dout2, N_EDGES, EDIM, HDIM);
}
